// SDT_20272245637441
// MI455X (gfx1250) — matrix operand order repaired and checked by reading, not compiled, not run on hardware
//
#include <hip/hip_runtime.h>

#define BATCH   8192
#define IN_DIM  256
#define HIDDEN  64
#define NODES   255
#define LEAVES  256

typedef __attribute__((ext_vector_type(16))) __bf16 v16bf;
typedef __attribute__((ext_vector_type(8)))  float  v8f;

union Frag {               // 8 dwords = 16 packed bf16 (pair-packed: low half = even K)
    unsigned u[8];
    uint4    q[2];
    v16bf    v;
};

// ---------- helpers ----------
__device__ __forceinline__ unsigned short f32_to_bf16_rne(float f) {
    unsigned u = __float_as_uint(f);
    unsigned r = u + 0x7FFFu + ((u >> 16) & 1u);   // round-to-nearest-even
    return (unsigned short)(r >> 16);
}
__device__ __forceinline__ float bf16_to_f32(unsigned short h) {
    return __uint_as_float(((unsigned)h) << 16);
}

// ---------- kernel 0: split f32 -> bf16 hi/lo ----------
__global__ void sdt_split_kernel(const float* __restrict__ src,
                                 unsigned short* __restrict__ hi,
                                 unsigned short* __restrict__ lo, int n) {
    int i = blockIdx.x * blockDim.x + threadIdx.x;
    if (i < n) {
        float f = src[i];
        unsigned short h = f32_to_bf16_rne(f);
        unsigned short l = f32_to_bf16_rne(f - bf16_to_f32(h));
        hi[i] = h; lo[i] = l;
    }
}

// ---------- kernel 1: fused X*W1^T -> relu -> .w2 -> sigmoid ----------
// Block = 4 waves = 64 batch rows x 1 node. All 4 waves share the node's B tile
// in LDS (4x less L2 traffic for W1); each wave owns a private 16-row A tile.
// xhi/xlo  : (BATCH, IN_DIM/2)  packed-pair uints
// w1hi/w1lo: (NODES*HIDDEN, IN_DIM/2) packed-pair uints
__global__ __launch_bounds__(128) void sdt_gemm_kernel(
    const unsigned* __restrict__ xhi, const unsigned* __restrict__ xlo,
    const unsigned* __restrict__ w1hi, const unsigned* __restrict__ w1lo,
    const float* __restrict__ b1, const float* __restrict__ w2,
    const float* __restrict__ b2, float* __restrict__ node_out) {
    // LDS, indexed in packed-pair dwords: B[h][k/2] (64x16) shared, A[m][k/2] per wave
    __shared__ __attribute__((aligned(16))) unsigned sBhi[64 * 16];
    __shared__ __attribute__((aligned(16))) unsigned sBlo[64 * 16];
    __shared__ __attribute__((aligned(16))) unsigned sAhi[4][16 * 16];
    __shared__ __attribute__((aligned(16))) unsigned sAlo[4][16 * 16];

    const int tid  = threadIdx.x;
    const int wave = tid >> 5;
    const int lane = tid & 31;
    const int half = lane >> 4;     // 0: lanes 0-15, 1: lanes 16-31
    const int l15  = lane & 15;
    const int M0   = blockIdx.x * 64 + wave * 16;   // this wave's batch tile base
    const int n    = blockIdx.y;                    // node index

    v8f zero = {0.f, 0.f, 0.f, 0.f, 0.f, 0.f, 0.f, 0.f};
    v8f acc[4];                          // 4 N-subtiles of 16 -> 64 hidden cols
#pragma unroll
    for (int t = 0; t < 4; ++t) acc[t] = zero;

    const unsigned* wBaseHi = w1hi + (size_t)n * HIDDEN * (IN_DIM / 2);
    const unsigned* wBaseLo = w1lo + (size_t)n * HIDDEN * (IN_DIM / 2);

    for (int kc = 0; kc < IN_DIM / 32; ++kc) {     // 8 chunks of K=32
        const int k0d = kc * 16;                   // dword (pair) column base
        // --- stage B cooperatively with all 128 threads: 2x uint4 per thread ---
#pragma unroll
        for (int j = 0; j < 2; ++j) {
            int d = (tid * 2 + j) * 4;             // flat dword 0..1020
            int h = d >> 4, c = d & 15;
            uint4 vh = *(const uint4*)(wBaseHi + (size_t)h * (IN_DIM / 2) + k0d + c);
            uint4 vl = *(const uint4*)(wBaseLo + (size_t)h * (IN_DIM / 2) + k0d + c);
            *(uint4*)(&sBhi[d]) = vh;
            *(uint4*)(&sBlo[d]) = vl;
            if (kc + 1 < IN_DIM / 32) {            // prefetch next chunk's W1 lines
                __builtin_prefetch(wBaseHi + (size_t)h * (IN_DIM / 2) + k0d + 16 + c, 0, 3);
                __builtin_prefetch(wBaseLo + (size_t)h * (IN_DIM / 2) + k0d + 16 + c, 0, 3);
            }
        }
        // --- stage this wave's A tile: 2x uint4 per lane per buffer ---
#pragma unroll
        for (int j = 0; j < 2; ++j) {
            int d = (lane * 2 + j) * 4;            // flat dword 0..252
            int m = d >> 4, c = d & 15;
            uint4 vh = *(const uint4*)(xhi + (size_t)(M0 + m) * (IN_DIM / 2) + k0d + c);
            uint4 vl = *(const uint4*)(xlo + (size_t)(M0 + m) * (IN_DIM / 2) + k0d + c);
            *(uint4*)(&sAhi[wave][d]) = vh;
            *(uint4*)(&sAlo[wave][d]) = vl;
        }
        __syncthreads();

        // --- A fragments per ISA 16-bit A 16x32 layout ---
        // lanes 0-15: M=lane, V0-3 K=0..7, V4-7 K=16..23 ; lanes 16-31: K=8..15 / 24..31
        Frag ahi, alo;
        {
            int b0  = l15 * 16 + half * 4;         // dwords for V0..V3
            int b1i = l15 * 16 + 8 + half * 4;     // dwords for V4..V7
            ahi.q[0] = *(const uint4*)(&sAhi[wave][b0]);
            ahi.q[1] = *(const uint4*)(&sAhi[wave][b1i]);
            alo.q[0] = *(const uint4*)(&sAlo[wave][b0]);
            alo.q[1] = *(const uint4*)(&sAlo[wave][b1i]);
        }
#pragma unroll
        for (int t = 0; t < 4; ++t) {
            // B fragments (32x16): lanes 0-15: N=lane,K=0..7/16..23 ; lanes 16-31: N=lane-16,K=8..15/24..31
            Frag bhi, blo;
            int hb = (t * 16 + l15) * 16 + half * 4;
            bhi.q[0] = *(const uint4*)(&sBhi[hb]);
            bhi.q[1] = *(const uint4*)(&sBhi[hb + 8]);
            blo.q[0] = *(const uint4*)(&sBlo[hb]);
            blo.q[1] = *(const uint4*)(&sBlo[hb + 8]);
            // x.w = xh.wh + xh.wl + xl.wh  (f32 accumulate)
            acc[t] = __builtin_amdgcn_wmma_f32_16x16x32_bf16(
                false, ahi.v, false, bhi.v, (short)0, acc[t], false, false);
            acc[t] = __builtin_amdgcn_wmma_f32_16x16x32_bf16(
                false, ahi.v, false, blo.v, (short)0, acc[t], false, false);
            acc[t] = __builtin_amdgcn_wmma_f32_16x16x32_bf16(
                false, alo.v, false, bhi.v, (short)0, acc[t], false, false);
        }
        __syncthreads();                           // protect LDS before next stage
    }

    // --- epilogue: +b1, relu, .w2, reduce over hidden, +b2, sigmoid ---
    float b1v[4], w2v[4];
#pragma unroll
    for (int t = 0; t < 4; ++t) {
        int h = t * 16 + l15;
        b1v[t] = b1[n * HIDDEN + h];
        w2v[t] = w2[n * HIDDEN + h];
    }
    float b2v = b2[n];
#pragma unroll
    for (int e = 0; e < 8; ++e) {       // C/D layout: M = e + 8*half, N = l15 (per subtile)
        float s = 0.f;
#pragma unroll
        for (int t = 0; t < 4; ++t) {
            float hv = acc[t][e] + b1v[t];
            hv = fmaxf(hv, 0.f);
            s += hv * w2v[t];
        }
        s += __shfl_xor(s, 1, 32);
        s += __shfl_xor(s, 2, 32);
        s += __shfl_xor(s, 4, 32);
        s += __shfl_xor(s, 8, 32);       // masks stay within each 16-lane half
        if (l15 == 0) {
            int m = e + 8 * half;
            float logit = s + b2v;
            float sig = 1.f / (1.f + __expf(-logit));
            node_out[(size_t)(M0 + m) * NODES + n] = sig;
        }
    }
}

// ---------- kernel 2: tree reach probabilities ----------
__global__ __launch_bounds__(64) void sdt_tree_kernel(
    const float* __restrict__ node_out, const float* __restrict__ leaf,
    float* __restrict__ h_out, float* __restrict__ path_probs,
    float* __restrict__ node_reach) {
    __shared__ float sp[64 * NODES];               // 63.75 KB
    const int tid = threadIdx.x;
    const int r0  = blockIdx.x * 64;
    const float* src = node_out + (size_t)r0 * NODES;
    for (int i = tid; i < 64 * NODES; i += 64) sp[i] = src[i];  // coalesced stage
    __syncthreads();

    const float* p = sp + tid * NODES;
    const int b = r0 + tid;
    float hsum = 0.f;
    for (int m = 1; m < 512; ++m) {                // heap index walk
        int lvl = 31 - __clz(m);
        float r = 1.f;
        for (int k = lvl; k >= 1; --k) {
            float pv = p[(m >> k) - 1];
            r *= ((m >> (k - 1)) & 1) ? pv : (1.f - pv);
        }
        if (m < 256) {
            node_reach[(size_t)b * NODES + (m - 1)] = r;
        } else {
            int j = m - 256;
            path_probs[(size_t)b * LEAVES + j] = r;
            hsum += r * leaf[j];
        }
    }
    h_out[b] = hsum;
}

// ---------- host ----------
extern "C" void kernel_launch(void* const* d_in, const int* in_sizes, int n_in,
                              void* d_out, int out_size, void* d_ws, size_t ws_size,
                              hipStream_t stream) {
    const float* x    = (const float*)d_in[0];   // (8192, 256)
    const float* w1   = (const float*)d_in[1];   // (255, 64, 256)
    const float* b1   = (const float*)d_in[2];   // (255, 64)
    const float* w2   = (const float*)d_in[3];   // (255, 64)
    const float* b2   = (const float*)d_in[4];   // (255,)
    const float* leaf = (const float*)d_in[5];   // (256, 1)

    float* out        = (float*)d_out;
    float* h_out      = out;                                     // 8192
    float* path_probs = out + BATCH;                             // 8192*256
    float* node_out   = path_probs + (size_t)BATCH * LEAVES;     // 8192*255
    float* node_reach = node_out + (size_t)BATCH * NODES;        // 8192*255

    const size_t nx = (size_t)BATCH * IN_DIM;            // 2,097,152
    const size_t nw = (size_t)NODES * HIDDEN * IN_DIM;   // 4,177,920
    unsigned short* xhi  = (unsigned short*)d_ws;
    unsigned short* xlo  = xhi + nx;
    unsigned short* w1hi = xlo + nx;
    unsigned short* w1lo = w1hi + nw;                    // total ~24 MB of ws

    sdt_split_kernel<<<(int)((nx + 255) / 256), 256, 0, stream>>>(x, xhi, xlo, (int)nx);
    sdt_split_kernel<<<(int)((nw + 255) / 256), 256, 0, stream>>>(w1, w1hi, w1lo, (int)nw);

    dim3 g1(BATCH / 64, NODES);
    sdt_gemm_kernel<<<g1, 128, 0, stream>>>(
        (const unsigned*)xhi, (const unsigned*)xlo,
        (const unsigned*)w1hi, (const unsigned*)w1lo,
        b1, w2, b2, node_out);

    sdt_tree_kernel<<<BATCH / 64, 64, 0, stream>>>(node_out, leaf, h_out,
                                                   path_probs, node_reach);
}
